// VRWKV_SpatialMix_50096498541307
// MI455X (gfx1250) — compile-verified, matrix operand order checked by reading, not run on hardware
//
#include <hip/hip_runtime.h>
#include <hip/hip_bf16.h>

// VRWKV SpatialMix for MI455X (gfx1250), wave32 + WMMA.
// B=8, H=W=64, C=256, T=4096 (B derived from in_sizes[0]).
// Workspace layout (floats), SZ = B*T*C:
//   slot0: xx            -> reused as scan temp A -> reused as LN output
//   slot1: k
//   slot2: v             -> reused as v2 (rwkv) output of pass 2
//   slot3: sr
//   slot4: v1
//   slot5: scan temp B
//   slot6: scan temp P
//   7*SZ : chunk-state block (12 arrays of B*NC*C floats, ~1.6 MB)
// Total ~7*32MB + 1.6MB ~= 236 MB.

#define Hc 64
#define Wc 64
#define Cc 256
#define Tc (Hc * Wc)
#define Lc 256           // scan chunk length
#define NCc (Tc / Lc)    // chunks per sequence = 16
#define NEGF (-1e30f)

typedef __bf16 v16bf __attribute__((ext_vector_type(16)));
typedef float  v8f   __attribute__((ext_vector_type(8)));

// fast reciprocal: v_rcp_f32 (~1 ulp) -- denominators here are >= O(1),
// well conditioned, so this is ample vs. the v_div_scale/fixup chain.
__device__ __forceinline__ float frcp(float x) { return __builtin_amdgcn_rcpf(x); }

// ---- sequence index map: pass 2 walks W-major order over the H-major tensor
__device__ __forceinline__ int seqmem(int s, int swap) {
  return swap ? (((s & (Hc - 1)) << 6) | (s >> 6)) : s;
}

// ---- one WKV recurrence step: state (a,b,p) <- append element (kt,vt)
__device__ __forceinline__ void wkv_step(float& a, float& b, float& p,
                                         float wc, float kt, float vt) {
  float pd = p - wc;
  float q  = fmaxf(pd, kt);
  float e1 = __expf(pd - q);
  float e2 = __expf(kt - q);
  a = e1 * a + e2 * vt;
  b = e1 * b + e2;
  p = q;
}

// ---- compose: S1 (older, already decayed to its segment end) + segment
// summary S2 of length L (Lw = L*wc). Associative in log space.
__device__ __forceinline__ void wkv_combine(float& a, float& b, float& p,
                                            float a2, float b2, float p2,
                                            float Lw) {
  float p1d = p - Lw;
  float q   = fmaxf(p1d, p2);
  float e1  = __expf(p1d - q);
  float e2  = __expf(p2 - q);
  a = e1 * a + e2 * a2;
  b = e1 * b + e2 * b2;
  p = q;
}

// =====================================================================
// Kernel 1: xx = a0*x + a1*dw1(x) + a2*dw3(x) + a3*dw5(x)  (depthwise)
// block = 256 threads (one channel each), grid = B*H, loop over W.
// =====================================================================
__global__ void k_conv(const float* __restrict__ x, const float* __restrict__ alpha,
                       const float* __restrict__ w1, const float* __restrict__ w3,
                       const float* __restrict__ w5, float* __restrict__ xx) {
  const int c  = threadIdx.x;
  const int b  = blockIdx.x / Hc;
  const int h  = blockIdx.x % Hc;
  const float a0 = alpha[0], a1 = alpha[1], a2 = alpha[2], a3 = alpha[3];

  float coef[25];
#pragma unroll
  for (int di = 0; di < 5; ++di) {
#pragma unroll
    for (int dj = 0; dj < 5; ++dj) {
      float cc = a3 * w5[c * 25 + di * 5 + dj];
      if (di >= 1 && di <= 3 && dj >= 1 && dj <= 3)
        cc += a2 * w3[c * 9 + (di - 1) * 3 + (dj - 1)];
      if (di == 2 && dj == 2) cc += a1 * w1[c] + a0;
      coef[di * 5 + dj] = cc;
    }
  }
  const float* xb  = x  + (size_t)b * Tc * Cc;
  float*       xxb = xx + (size_t)b * Tc * Cc;
  for (int w = 0; w < Wc; ++w) {
    float acc = 0.f;
#pragma unroll
    for (int di = 0; di < 5; ++di) {
      const int hh = h + di - 2;
      if (hh < 0 || hh >= Hc) continue;
#pragma unroll
      for (int dj = 0; dj < 5; ++dj) {
        const int ww = w + dj - 2;
        if (ww < 0 || ww >= Wc) continue;
        acc += coef[di * 5 + dj] * xb[(size_t)(hh * Wc + ww) * Cc + c];
      }
    }
    xxb[(size_t)(h * Wc + w) * Cc + c] = acc;
  }
}

// =====================================================================
// Kernel 2: fused mix + triple GEMM (k, v, sigmoid(r)).
//   Y = A @ W^T, A formed on the fly: a = xx + m*(x-xx); the (x-xx)
//   delta is computed once and shared by the k/v/r chains (1 FMA each).
// mk/mv/mr are staged in LDS once per block (3 KB) -> one 32-bit ds
// address instead of three global streams. 32-bit offsets let loads use
// the saddr + voffset form (no 64-bit per-lane pointers, no spills).
// One wave per 16x16 output tile; block = 128 (4 waves),
// grid = (M/64, C/16). Fragment layouts per CDNA5 ISA 7.12.2.
// =====================================================================
__global__ void k_gemm_kvr(const float* __restrict__ x,  const float* __restrict__ xx,
                           const float* __restrict__ mk_, const float* __restrict__ mv_,
                           const float* __restrict__ mr_,
                           const float* __restrict__ Wk, const float* __restrict__ Wv,
                           const float* __restrict__ Wr,
                           float* __restrict__ kO, float* __restrict__ vO,
                           float* __restrict__ srO) {
  __shared__ float smix[3 * Cc];
  for (int i = threadIdx.x; i < 3 * Cc; i += 128) {
    smix[i] = (i < Cc) ? mk_[i] : (i < 2 * Cc ? mv_[i - Cc] : mr_[i - 2 * Cc]);
  }
  __syncthreads();

  const int lane  = threadIdx.x & 31;
  const int wave  = threadIdx.x >> 5;
  const int mtile = blockIdx.x * 4 + wave;
  const int ntile = blockIdx.y;
  const int half  = lane >> 4;
  const int l16   = lane & 15;
  const unsigned ao = (unsigned)(mtile * 16 + l16) * Cc;  // A offset (x, xx)
  const unsigned bo = (unsigned)(ntile * 16 + l16) * Cc;  // B offset (Wk/Wv/Wr)

  v8f accK = {}, accV = {}, accR = {};

#pragma unroll 2
  for (int kb = 0; kb < Cc; kb += 32) {
    v16bf aK, aV, aR, bK, bV, bR;
#pragma unroll
    for (int j = 0; j < 16; ++j) {
      const int kk = kb + ((j >> 3) << 4) + (half << 3) + (j & 7);
      const float xv  = x[ao + kk];
      const float xxv = xx[ao + kk];
      const float d   = xv - xxv;                   // shared delta
      aK[j] = (__bf16)(xxv + smix[kk] * d);         // 1 FMA per chain
      aV[j] = (__bf16)(xxv + smix[Cc + kk] * d);
      aR[j] = (__bf16)(xxv + smix[2 * Cc + kk] * d);
      bK[j] = (__bf16)Wk[bo + kk];
      bV[j] = (__bf16)Wv[bo + kk];
      bR[j] = (__bf16)Wr[bo + kk];
    }
    accK = __builtin_amdgcn_wmma_f32_16x16x32_bf16(false, aK, false, bK, (short)0, accK, false, false);
    accV = __builtin_amdgcn_wmma_f32_16x16x32_bf16(false, aV, false, bV, (short)0, accV, false, false);
    accR = __builtin_amdgcn_wmma_f32_16x16x32_bf16(false, aR, false, bR, (short)0, accR, false, false);
  }
#pragma unroll
  for (int r = 0; r < 8; ++r) {
    const unsigned o = (unsigned)(mtile * 16 + (half << 3) + r) * Cc
                     + (unsigned)(ntile * 16 + l16);
    kO[o]  = accK[r];
    vO[o]  = accV[r];
    srO[o] = frcp(1.f + __expf(-accR[r]));   // sigmoid via v_rcp_f32
  }
}

// =====================================================================
// WKV level 1: per-(b,chunk,c) forward & backward chunk summaries.
// chunk-state block cb: 12 arrays of N=B*NC*C floats:
//   [0..2]=sumF(a,b,p) [3..5]=sumB [6..8]=entF [9..11]=entB
// =====================================================================
__global__ void k_wkv_chunks(const float* __restrict__ kT, const float* __restrict__ vT,
                             const float* __restrict__ sd, int pass, int swap,
                             float* __restrict__ cb, int B) {
  const int tid = blockIdx.x * blockDim.x + threadIdx.x;
  const int c  = tid % Cc;
  const int ch = (tid / Cc) % NCc;
  const int b  = tid / (Cc * NCc);
  if (b >= B) return;
  const float wc = sd[pass * Cc + c] * (1.f / Tc);
  const float* kb = kT + (size_t)b * Tc * Cc + c;
  const float* vb = vT + (size_t)b * Tc * Cc + c;
  const size_t N  = (size_t)B * NCc * Cc;
  const size_t ci = ((size_t)b * NCc + ch) * Cc + c;

  float a = 0.f, bb = 0.f, p = NEGF;
  for (int i = 0; i < Lc; ++i) {
    const int t = seqmem(ch * Lc + i, swap);
    wkv_step(a, bb, p, wc, kb[(size_t)t * Cc], vb[(size_t)t * Cc]);
  }
  cb[0 * N + ci] = a; cb[1 * N + ci] = bb; cb[2 * N + ci] = p;

  a = 0.f; bb = 0.f; p = NEGF;
  for (int i = Lc - 1; i >= 0; --i) {
    const int t = seqmem(ch * Lc + i, swap);
    wkv_step(a, bb, p, wc, kb[(size_t)t * Cc], vb[(size_t)t * Cc]);
  }
  cb[3 * N + ci] = a; cb[4 * N + ci] = bb; cb[5 * N + ci] = p;
}

// =====================================================================
// WKV level 2: per-(b,c) 16-step scan of chunk summaries -> exclusive
// entry states for each chunk, both directions.
// =====================================================================
__global__ void k_wkv_entries(const float* __restrict__ sd, int pass,
                              float* __restrict__ cb, int B) {
  const int tid = blockIdx.x * blockDim.x + threadIdx.x;
  const int c = tid % Cc;
  const int b = tid / Cc;
  if (b >= B) return;
  const float wc = sd[pass * Cc + c] * (1.f / Tc);
  const float Lw = (float)Lc * wc;
  const size_t N = (size_t)B * NCc * Cc;

  float a = 0.f, bb = 0.f, p = NEGF;
  for (int ch = 0; ch < NCc; ++ch) {
    const size_t ci = ((size_t)b * NCc + ch) * Cc + c;
    cb[6 * N + ci] = a; cb[7 * N + ci] = bb; cb[8 * N + ci] = p;
    wkv_combine(a, bb, p, cb[0 * N + ci], cb[1 * N + ci], cb[2 * N + ci], Lw);
  }
  a = 0.f; bb = 0.f; p = NEGF;
  for (int ch = NCc - 1; ch >= 0; --ch) {
    const size_t ci = ((size_t)b * NCc + ch) * Cc + c;
    cb[9 * N + ci] = a; cb[10 * N + ci] = bb; cb[11 * N + ci] = p;
    wkv_combine(a, bb, p, cb[3 * N + ci], cb[4 * N + ci], cb[5 * N + ci], Lw);
  }
}

// =====================================================================
// WKV level 3: per-(b,chunk,c): backward sweep stores exclusive suffix
// states (own chunk only -> no cross-thread hazard), forward sweep
// combines and writes y at the element's memory position.
// =====================================================================
__global__ void k_wkv_apply(const float* __restrict__ kT, const float* __restrict__ vT,
                            const float* __restrict__ sd, const float* __restrict__ sf,
                            int pass, int swap, const float* __restrict__ cb,
                            float* __restrict__ tA, float* __restrict__ tB,
                            float* __restrict__ tP, float* __restrict__ yO, int B) {
  const int tid = blockIdx.x * blockDim.x + threadIdx.x;
  const int c  = tid % Cc;
  const int ch = (tid / Cc) % NCc;
  const int b  = tid / (Cc * NCc);
  if (b >= B) return;
  const float wc = sd[pass * Cc + c] * (1.f / Tc);
  const float uc = sf[pass * Cc + c] * (1.f / Tc);
  const size_t N  = (size_t)B * NCc * Cc;
  const size_t ci = ((size_t)b * NCc + ch) * Cc + c;
  const size_t boff = (size_t)b * Tc * Cc + c;
  const float* kb = kT + boff;
  const float* vb = vT + boff;
  float* tAb = tA + boff;
  float* tBb = tB + boff;
  float* tPb = tP + boff;
  float* yb  = yO + boff;

  // backward sweep: store exclusive suffix state at each s in this chunk
  float a = cb[9 * N + ci], bb = cb[10 * N + ci], p = cb[11 * N + ci];
  for (int i = Lc - 1; i >= 0; --i) {
    const int s = ch * Lc + i;
    tAb[(size_t)s * Cc] = a;
    tBb[(size_t)s * Cc] = bb;
    tPb[(size_t)s * Cc] = p;
    const int t = seqmem(s, swap);
    wkv_step(a, bb, p, wc, kb[(size_t)t * Cc], vb[(size_t)t * Cc]);
  }
  // forward sweep: combine exclusive prefix + suffix + self
  a = cb[6 * N + ci]; bb = cb[7 * N + ci]; p = cb[8 * N + ci];
  for (int i = 0; i < Lc; ++i) {
    const int s = ch * Lc + i;
    const int t = seqmem(s, swap);
    const float kt = kb[(size_t)t * Cc];
    const float vt = vb[(size_t)t * Cc];
    const float ab = tAb[(size_t)s * Cc];
    const float b2 = tBb[(size_t)s * Cc];
    const float pb = tPb[(size_t)s * Cc];
    const float pc = uc + kt;
    const float q  = fmaxf(fmaxf(p, pb), pc);
    const float ef = __expf(p - q);
    const float eb = __expf(pb - q);
    const float ec = __expf(pc - q);
    const float num = ef * a + eb * ab + ec * vt;
    const float den = ef * bb + eb * b2 + ec;
    yb[(size_t)t * Cc] = num * frcp(den);    // v_rcp_f32, not div chain
    wkv_step(a, bb, p, wc, kt, vt);
  }
}

// =====================================================================
// LayerNorm over C: one wave per row, shfl_xor reduction (wave32).
// =====================================================================
__global__ void k_ln(const float* __restrict__ in, const float* __restrict__ g,
                     const float* __restrict__ bt, float* __restrict__ out, int rows) {
  const int lane = threadIdx.x & 31;
  const int wave = threadIdx.x >> 5;
  const int row  = blockIdx.x * 8 + wave;
  if (row >= rows) return;
  const float* r = in + (size_t)row * Cc;
  float s = 0.f, ss = 0.f, vals[8];
#pragma unroll
  for (int i = 0; i < 8; ++i) {
    const float v = r[lane + 32 * i];
    vals[i] = v; s += v; ss += v * v;
  }
#pragma unroll
  for (int o = 16; o > 0; o >>= 1) {
    s  += __shfl_xor(s, o, 32);
    ss += __shfl_xor(ss, o, 32);
  }
  const float mu = s * (1.f / Cc);
  const float rs = rsqrtf(ss * (1.f / Cc) - mu * mu + 1e-5f);
  float* w = out + (size_t)row * Cc;
#pragma unroll
  for (int i = 0; i < 8; ++i) {
    const int cix = lane + 32 * i;
    w[cix] = (vals[i] - mu) * rs * g[cix] + bt[cix];
  }
}

// =====================================================================
// Final GEMM: out = (sr * rwkv_ln) @ Wo^T  (WMMA bf16, f32 acc)
// =====================================================================
__global__ void k_gemm_out(const float* __restrict__ srA, const float* __restrict__ lnA,
                           const float* __restrict__ Wo, float* __restrict__ out) {
  const int lane  = threadIdx.x & 31;
  const int wave  = threadIdx.x >> 5;
  const int mtile = blockIdx.x * 4 + wave;
  const int ntile = blockIdx.y;
  const int half  = lane >> 4;
  const int l16   = lane & 15;
  const unsigned ao = (unsigned)(mtile * 16 + l16) * Cc;
  const unsigned bo = (unsigned)(ntile * 16 + l16) * Cc;

  v8f acc = {};
#pragma unroll 2
  for (int kb = 0; kb < Cc; kb += 32) {
    if (kb + 32 < Cc) {
      __builtin_prefetch(&srA[ao + kb + 32], 0, 1);
      __builtin_prefetch(&lnA[ao + kb + 32], 0, 1);
    }
    v16bf aF, bF;
#pragma unroll
    for (int j = 0; j < 16; ++j) {
      const int kk = kb + ((j >> 3) << 4) + (half << 3) + (j & 7);
      aF[j] = (__bf16)(srA[ao + kk] * lnA[ao + kk]);
      bF[j] = (__bf16)Wo[bo + kk];
    }
    acc = __builtin_amdgcn_wmma_f32_16x16x32_bf16(false, aF, false, bF, (short)0, acc, false, false);
  }
#pragma unroll
  for (int r = 0; r < 8; ++r) {
    const unsigned o = (unsigned)(mtile * 16 + (half << 3) + r) * Cc
                     + (unsigned)(ntile * 16 + l16);
    out[o] = acc[r];
  }
}

extern "C" void kernel_launch(void* const* d_in, const int* in_sizes, int n_in,
                              void* d_out, int out_size, void* d_ws, size_t ws_size,
                              hipStream_t stream) {
  const float* x     = (const float*)d_in[0];
  const float* mixk  = (const float*)d_in[1];
  const float* mixv  = (const float*)d_in[2];
  const float* mixr  = (const float*)d_in[3];
  const float* sd    = (const float*)d_in[4];   // spatial_decay [2,C]
  const float* sf    = (const float*)d_in[5];   // spatial_first [2,C]
  const float* alpha = (const float*)d_in[6];
  const float* w1    = (const float*)d_in[7];
  const float* w3    = (const float*)d_in[8];
  const float* w5    = (const float*)d_in[9];
  const float* Wk    = (const float*)d_in[10];
  const float* Wv    = (const float*)d_in[11];
  const float* Wr    = (const float*)d_in[12];
  const float* Wo    = (const float*)d_in[13];
  const float* lng   = (const float*)d_in[14];
  const float* lnb   = (const float*)d_in[15];

  const int B = in_sizes[0] / (Tc * Cc);
  const int M = B * Tc;
  const size_t SZ = (size_t)B * Tc * Cc;

  float* ws    = (float*)d_ws;
  float* xx    = ws;                 // slot0
  float* kbuf  = ws + 1 * SZ;        // slot1
  float* vbuf  = ws + 2 * SZ;        // slot2 (later: v2 / rwkv)
  float* srbuf = ws + 3 * SZ;        // slot3
  float* v1    = ws + 4 * SZ;        // slot4
  float* tB_   = ws + 5 * SZ;        // slot5
  float* tP_   = ws + 6 * SZ;        // slot6
  float* cb    = ws + 7 * SZ;        // chunk states (12 * B*NC*C floats)
  float* tA_   = xx;                 // xx dead after k_gemm_kvr
  float* lnbuf = xx;                 // temps dead after pass-2 apply

  const int scanThr = B * NCc * Cc;                 // 32768 for B=8
  const int scanBlk = (scanThr + 255) / 256;
  const int entBlk  = (B * Cc + 255) / 256;

  // 1) depthwise conv blend
  k_conv<<<B * Hc, 256, 0, stream>>>(x, alpha, w1, w3, w5, xx);
  // 2) fused mix + k/v/r GEMMs (WMMA)
  k_gemm_kvr<<<dim3(M / 64, Cc / 16), 128, 0, stream>>>(
      x, xx, mixk, mixv, mixr, Wk, Wv, Wr, kbuf, vbuf, srbuf);
  // 3) bi-WKV pass 1 (H-major, decay/first row 0): k,v -> v1
  k_wkv_chunks <<<scanBlk, 256, 0, stream>>>(kbuf, vbuf, sd, 0, 0, cb, B);
  k_wkv_entries<<<entBlk, 256, 0, stream>>>(sd, 0, cb, B);
  k_wkv_apply  <<<scanBlk, 256, 0, stream>>>(kbuf, vbuf, sd, sf, 0, 0, cb,
                                             tA_, tB_, tP_, v1, B);
  // 4) bi-WKV pass 2 (W-major via index map, row 1): k,v1 -> vbuf (=rwkv)
  k_wkv_chunks <<<scanBlk, 256, 0, stream>>>(kbuf, v1, sd, 1, 1, cb, B);
  k_wkv_entries<<<entBlk, 256, 0, stream>>>(sd, 1, cb, B);
  k_wkv_apply  <<<scanBlk, 256, 0, stream>>>(kbuf, v1, sd, sf, 1, 1, cb,
                                             tA_, tB_, tP_, vbuf, B);
  // 5) LayerNorm over C
  k_ln<<<(M + 7) / 8, 256, 0, stream>>>(vbuf, lng, lnb, lnbuf, M);
  // 6) gated output GEMM (WMMA)
  k_gemm_out<<<dim3(M / 64, Cc / 16), 128, 0, stream>>>(srbuf, lnbuf, Wo,
                                                        (float*)d_out);
}
